// AttentionBlock_82025285419402
// MI455X (gfx1250) — matrix operand order repaired and checked by reading, not compiled, not run on hardware
//
#include <hip/hip_runtime.h>
#include <stdint.h>

// Problem constants (reference: B=8, C=512, T=2048, K=V=512)
#define B_ 8
#define C_ 512
#define T_ 2048
#define K_ 512
#define LOG2E 1.4426950408889634f
#define SCALE 0.04419417382415922f   // 1/sqrt(512)

typedef __attribute__((ext_vector_type(16))) __bf16 v16bf;
typedef __attribute__((ext_vector_type(8)))  float  v8f;
typedef unsigned int u32x4 __attribute__((ext_vector_type(4)));
typedef int          i32x4 __attribute__((ext_vector_type(4)));
typedef int          i32x8 __attribute__((ext_vector_type(8)));

__device__ __forceinline__ uint16_t f2bf(float f) {
  union { float f; uint32_t u; } c; c.f = f;
  uint32_t u = c.u;
  uint32_t r = u + 0x7FFFu + ((u >> 16) & 1u);   // round-to-nearest-even
  return (uint16_t)(r >> 16);
}

union FragBF { v16bf v; uint4 q[2]; uint16_t h[16]; };

// A-matrix 16x32 bf16 fragment from row-major source (stride in elements).
// ISA layout: lane<16 -> row=lane, K = {0..7, 16..23}; lane>=16 -> row=lane-16, K = {8..15, 24..31}
__device__ __forceinline__ v16bf load_a_bf16(const uint16_t* base, int stride, int lane) {
  const int half = lane >> 4, m = lane & 15;
  const uint16_t* p = base + m * stride + half * 8;
  FragBF f;
  f.q[0] = *reinterpret_cast<const uint4*>(p);        // K 0..7   (+8 if half)
  f.q[1] = *reinterpret_cast<const uint4*>(p + 16);   // K 16..23 (+8 if half)
  return f.v;
}

// B-matrix 32x16 bf16 fragment; source row n supplies column n (row-major, stride elems).
// ISA layout: lane<16 -> col=lane, K = {0..7, 16..23}; lane>=16 -> col=lane-16, K = {8..15, 24..31}
__device__ __forceinline__ v16bf load_b_bf16(const uint16_t* base, int stride, int lane) {
  const int half = lane >> 4, n = lane & 15;
  const uint16_t* p = base + n * stride + half * 8;
  FragBF f;
  f.q[0] = *reinterpret_cast<const uint4*>(p);
  f.q[1] = *reinterpret_cast<const uint4*>(p + 16);
  return f.v;
}

// A fragment from fp32 row-major source, converted to bf16 on the fly.
__device__ __forceinline__ v16bf load_a_f32(const float* base, int stride, int lane) {
  const int half = lane >> 4, m = lane & 15;
  const float* p = base + m * stride + half * 8;
  const float4* p4 = reinterpret_cast<const float4*>(p);
  float4 g0 = p4[0], g1 = p4[1], g2 = p4[4], g3 = p4[5];  // K 0..7 and 16..23 (+8 if half)
  FragBF f;
  f.h[0] = f2bf(g0.x); f.h[1] = f2bf(g0.y); f.h[2]  = f2bf(g0.z); f.h[3]  = f2bf(g0.w);
  f.h[4] = f2bf(g1.x); f.h[5] = f2bf(g1.y); f.h[6]  = f2bf(g1.z); f.h[7]  = f2bf(g1.w);
  f.h[8] = f2bf(g2.x); f.h[9] = f2bf(g2.y); f.h[10] = f2bf(g2.z); f.h[11] = f2bf(g2.w);
  f.h[12]= f2bf(g3.x); f.h[13]= f2bf(g3.y); f.h[14] = f2bf(g3.z); f.h[15] = f2bf(g3.w);
  return f.v;
}

__device__ __forceinline__ v8f wmma_bf16(v16bf a, v16bf b, v8f c) {
  return __builtin_amdgcn_wmma_f32_16x16x32_bf16(false, a, false, b, (short)0, c, false, false);
}

// TDM: stage a [16 x 512] bf16 tile (one 2-D tensor slice) from global into LDS.
// D# built per CDNA5 ISA 8.3/8.4 (group0: count/lds/global/type, group1: dims+strides).
__device__ __forceinline__ void tdm_load_ktile(uint32_t lds_off, uint64_t gaddr) {
  u32x4 g0;
  g0[0] = 1u;                                        // count=1, user mode, no gather
  g0[1] = lds_off;                                   // lds_addr  [63:32]
  g0[2] = (uint32_t)gaddr;                           // global_addr[95:64] = addr[31:0]
  g0[3] = ((uint32_t)(gaddr >> 32) & 0x01FFFFFFu)    // addr[56:32]
        | 0x80000000u;                               // type=2 ("image") at [127:126]
  i32x8 g1;
  g1[0] = 0x00010000;                                // workgroup_mask=0, data_size=1 (2B)
  g1[1] = (int)(512u << 16);                         // tensor_dim0 = 512 (low 16)
  g1[2] = (int)(2048u << 16);                        // tensor_dim1 = 2048 (low 16)
  g1[3] = (int)(512u << 16);                         // tile_dim0 = 512
  g1[4] = 16;                                        // tile_dim1 = 16 rows
  g1[5] = 512;                                       // tensor_dim0_stride = 512
  g1[6] = 0;
  g1[7] = 0;
  i32x4 z4 = {0, 0, 0, 0};
#if __clang_major__ >= 23
  i32x8 z8 = {0, 0, 0, 0, 0, 0, 0, 0};
  __builtin_amdgcn_tensor_load_to_lds(g0, g1, z4, z4, z8, 0);
#else
  __builtin_amdgcn_tensor_load_to_lds(g0, g1, z4, z4, 0);
#endif
}

// ---------------- Kernel 1: x[B,C,T] fp32 -> xt[B,T,C] bf16 (LDS tiled transpose) ----
__global__ void k_transpose_x(const float* __restrict__ x, uint16_t* __restrict__ xt) {
  __shared__ float tile[32][33];
  const int b  = blockIdx.z;
  const int t0 = blockIdx.x * 32;
  const int c0 = blockIdx.y * 32;
  for (int i = threadIdx.y; i < 32; i += 8)
    tile[i][threadIdx.x] = x[(size_t)b * C_ * T_ + (size_t)(c0 + i) * T_ + t0 + threadIdx.x];
  __syncthreads();
  for (int i = threadIdx.y; i < 32; i += 8)
    xt[(size_t)b * T_ * C_ + (size_t)(t0 + i) * C_ + c0 + threadIdx.x] = f2bf(tile[threadIdx.x][i]);
}

// ---------------- Kernel 2: Wq/Wk/Wv fp32 -> bf16 ------------------------------------
__global__ void k_convert_w(const float* __restrict__ wq, const float* __restrict__ wk,
                            const float* __restrict__ wv, uint16_t* __restrict__ o) {
  const int i = blockIdx.x * 256 + threadIdx.x;          // 0 .. 512*512-1
  o[i]                 = f2bf(wq[i]);
  o[i + K_ * C_]       = f2bf(wk[i]);
  o[i + 2 * K_ * C_]   = f2bf(wv[i]);
}

// ---------------- Kernel 3: QKV projection (bf16 WMMA) -------------------------------
// grid: (M-tiles/4, K_/16, 3), block 128 (4 waves). One 16x16 tile per wave.
__global__ void k_qkv_gemm(const uint16_t* __restrict__ xt, const uint16_t* __restrict__ wbf,
                           const float* __restrict__ bq, const float* __restrict__ bk,
                           const float* __restrict__ bv,
                           uint16_t* __restrict__ qo, uint16_t* __restrict__ ko,
                           uint16_t* __restrict__ vT) {
  const int lane = threadIdx.x & 31;
  const int wave = threadIdx.x >> 5;
  const int tileM = blockIdx.x * 4 + wave;               // over B*T/16 = 1024
  const int tileN = blockIdx.y;                          // K_/16 = 32
  const int mat   = blockIdx.z;                          // 0=q, 1=k, 2=v

  const uint16_t* W   = wbf + (size_t)mat * K_ * C_;
  const float* bias   = (mat == 0) ? bq : (mat == 1) ? bk : bv;
  const uint16_t* abase = xt + (size_t)tileM * 16 * C_;
  const uint16_t* bbase = W  + (size_t)tileN * 16 * C_;

  v8f acc = {};
  #pragma unroll 4
  for (int kk = 0; kk < C_; kk += 32) {
    __builtin_prefetch(abase + kk + 64, 0, 3);
    __builtin_prefetch(bbase + kk + 64, 0, 3);
    v16bf a = load_a_bf16(abase + kk, C_, lane);
    v16bf b = load_b_bf16(bbase + kk, C_, lane);
    acc = wmma_bf16(a, b, acc);
  }

  const int n  = tileN * 16 + (lane & 15);
  const float bs = bias[n];
  #pragma unroll
  for (int i = 0; i < 8; ++i) {
    const int mg = tileM * 16 + i + (lane >> 4) * 8;     // global row in [0, B*T)
    const uint16_t val = f2bf(acc[i] + bs);
    if (mat == 0)      qo[(size_t)mg * K_ + n] = val;
    else if (mat == 1) ko[(size_t)mg * K_ + n] = val;
    else {
      const int b = mg >> 11, t = mg & (T_ - 1);         // T_ = 2048
      vT[(size_t)b * K_ * T_ + (size_t)n * T_ + t] = val; // v stored [B, V, T]
    }
  }
}

// ---------------- Kernel 4: scores = (q @ k^T) * SCALE -------------------------------
// grid: (T/16/4, T/16, B), block 128 (4 waves). The 16x512 k-tile (B operand) is staged
// once per block into LDS by the Tensor Data Mover; waves read B frags via ds_load_b128.
__global__ void k_scores_gemm(const uint16_t* __restrict__ qbf, const uint16_t* __restrict__ kbf,
                              float* __restrict__ w) {
  __shared__ uint16_t ktile[16 * K_];                    // 16 KB
  const int lane = threadIdx.x & 31;
  const int wave = threadIdx.x >> 5;
  const int tileT = blockIdx.y;                          // 0..127
  const int tileQ0 = blockIdx.x * 4;                     // 0..124
  const int b     = blockIdx.z;

  if (tileQ0 + 3 < tileT) return;                        // entire block masked (uniform)

  if (wave == 0) {
    const uint64_t gaddr = (uint64_t)(uintptr_t)(kbf + ((size_t)b * T_ + tileT * 16) * K_);
    const uint32_t loff  = (uint32_t)(uintptr_t)&ktile[0];  // flat->LDS offset (low 32 bits)
    tdm_load_ktile(loff, gaddr);
    __builtin_amdgcn_s_wait_tensorcnt(0);
  }
  __syncthreads();

  const int tileQ = tileQ0 + wave;
  if (tileQ >= tileT) {                                  // wave-uniform causal skip
    const uint16_t* abase = qbf + ((size_t)b * T_ + tileQ * 16) * K_;
    v8f acc = {};
    #pragma unroll 4
    for (int kk = 0; kk < K_; kk += 32) {
      __builtin_prefetch(abase + kk + 64, 0, 3);
      v16bf a = load_a_bf16(abase + kk, K_, lane);
      v16bf bfr = load_b_bf16(&ktile[kk], K_, lane);     // from LDS
      acc = wmma_bf16(a, bfr, acc);
    }

    const int ti = tileT * 16 + (lane & 15);
    float* out = w + (size_t)b * T_ * T_;
    #pragma unroll
    for (int i = 0; i < 8; ++i) {
      const int qi = tileQ * 16 + i + (lane >> 4) * 8;
      out[(size_t)qi * T_ + ti] = acc[i] * SCALE;        // masked elems overwritten to 0 later
    }
  }
}

// ---------------- Kernel 5: column softmax over query axis, in place ------------------
// For each key column t: softmax over q in [t, T); q<t -> exact 0 (causal mask in w).
__global__ void k_col_softmax(float* __restrict__ w) {
  const int t = blockIdx.x * 256 + threadIdx.x;
  const int b = blockIdx.y;
  float* col = w + (size_t)b * T_ * T_ + t;

  // Pass 1: online max + sum over valid q >= t (coalesced: contiguous t per row).
  float m = -3.4e38f, s = 0.0f;
  for (int q = t; q < T_; ++q) {
    const float v = col[(size_t)q * T_];
    if (v > m) { s *= exp2f((m - v) * LOG2E); m = v; }
    s += exp2f((v - m) * LOG2E);
  }
  const float inv = 1.0f / s;

  // Pass 2: normalize; write exact zeros into the masked region.
  for (int q = 0; q < T_; ++q) {
    float* p = col + (size_t)q * T_;
    const float o = (q < t) ? 0.0f : exp2f((*p - m) * LOG2E) * inv;
    *p = o;
  }
}

// ---------------- Kernel 6: out = w @ v, stored transposed [B, V, T] ------------------
// grid: (T/16/4, V/16, B), block 128. K-loop truncated at diagonal (w==0 beyond).
__global__ void k_out_gemm(const float* __restrict__ w, const uint16_t* __restrict__ vT,
                           float* __restrict__ out) {
  const int lane = threadIdx.x & 31;
  const int wave = threadIdx.x >> 5;
  const int tileQ = blockIdx.x * 4 + wave;               // 0..127
  const int tileN = blockIdx.y;                          // 0..31 (V/16)
  const int b     = blockIdx.z;

  const float*    abase = w  + (size_t)b * T_ * T_ + (size_t)tileQ * 16 * T_;
  const uint16_t* bbase = vT + (size_t)b * K_ * T_ + (size_t)tileN * 16 * T_;

  const int kend = ((tileQ + 1) * 16 + 31) & ~31;        // cover diagonal, step-32 aligned

  v8f acc = {};
  #pragma unroll 2
  for (int kk = 0; kk < kend; kk += 32) {
    __builtin_prefetch(abase + kk + 64, 0, 3);
    __builtin_prefetch(bbase + kk + 64, 0, 3);
    v16bf a = load_a_f32(abase + kk, T_, lane);          // fp32 probs -> bf16
    v16bf bfr = load_b_bf16(bbase + kk, T_, lane);
    acc = wmma_bf16(a, bfr, acc);
  }

  const int n = tileN * 16 + (lane & 15);
  #pragma unroll
  for (int i = 0; i < 8; ++i) {
    const int qg = tileQ * 16 + i + (lane >> 4) * 8;
    out[(size_t)b * K_ * T_ + (size_t)n * T_ + qg] = acc[i];   // [B, V, T]
  }
}

// -------------------------------------------------------------------------------------
extern "C" void kernel_launch(void* const* d_in, const int* in_sizes, int n_in,
                              void* d_out, int out_size, void* d_ws, size_t ws_size,
                              hipStream_t stream) {
  (void)in_sizes; (void)n_in; (void)out_size; (void)ws_size;

  const float* x  = (const float*)d_in[0];
  const float* Wq = (const float*)d_in[1];
  const float* bq = (const float*)d_in[2];
  const float* Wk = (const float*)d_in[3];
  const float* bk = (const float*)d_in[4];
  const float* Wv = (const float*)d_in[5];
  const float* bv = (const float*)d_in[6];

  float* out  = (float*)d_out;                               // [B, V, T]
  float* wout = out + (size_t)B_ * K_ * T_;                  // [B, T, T]

  // Workspace layout (bf16 elements): total ~65.5 MB
  uint16_t* xt  = (uint16_t*)d_ws;                           // [B*T, C]
  uint16_t* wbf = xt  + (size_t)B_ * T_ * C_;                // [3, K, C]
  uint16_t* qbf = wbf + (size_t)3 * K_ * C_;                 // [B*T, K]
  uint16_t* kbf = qbf + (size_t)B_ * T_ * K_;                // [B*T, K]
  uint16_t* vT  = kbf + (size_t)B_ * T_ * K_;                // [B, V, T]

  k_transpose_x<<<dim3(T_ / 32, C_ / 32, B_), dim3(32, 8), 0, stream>>>(x, xt);
  k_convert_w <<<dim3((K_ * C_) / 256), 256, 0, stream>>>(Wq, Wk, Wv, wbf);
  k_qkv_gemm  <<<dim3((B_ * T_ / 16) / 4, K_ / 16, 3), 128, 0, stream>>>(
                  xt, wbf, bq, bk, bv, qbf, kbf, vT);
  k_scores_gemm<<<dim3((T_ / 16) / 4, T_ / 16, B_), 128, 0, stream>>>(qbf, kbf, wout);
  k_col_softmax<<<dim3(T_ / 256, B_), 256, 0, stream>>>(wout);
  k_out_gemm  <<<dim3((T_ / 16) / 4, K_ / 16, B_), 128, 0, stream>>>(wout, vT, out);
}
